// multiBartDecoder_4054449127735
// MI455X (gfx1250) — compile-verified, matrix operand order checked by reading, not run on hardware
//
#include <hip/hip_runtime.h>
#include <hip/hip_bf16.h>

typedef __attribute__((ext_vector_type(16))) __bf16         v16bf;
typedef __attribute__((ext_vector_type(16))) unsigned short v16us;
typedef __attribute__((ext_vector_type(8)))  float          v8f;

#define L_    3
#define D_    1024
#define H_    16
#define HD_   64
#define FF_   4096
#define B_    2
#define T_    512
#define S_    512
#define BH_   (B_ * H_)
#define M_    (T_ * B_)
#define POFF_ 2
#define EPS_  1e-5f

#define USE_ASYNC_STAGE 1   // gfx1250 global_load_async_to_lds_b128 staging

// ------------------------------------------------------------- bf16 helpers
__device__ __forceinline__ unsigned short f2bf(float f) {
  return (unsigned short)((__float_as_uint(f) + 0x8000u) >> 16);
}
__device__ __forceinline__ unsigned int pk2bf(float a, float b) {
  return ((__float_as_uint(a) + 0x8000u) >> 16) |
         ((__float_as_uint(b) + 0x8000u) & 0xFFFF0000u);
}

// ------------------------------------------------------------- f32 -> bf16
__global__ __launch_bounds__(256)
void cvt_bf16_kernel(const float* __restrict__ src, unsigned short* __restrict__ dst,
                     int n4) {
  int i = blockIdx.x * 256 + threadIdx.x;
  if (i < n4) {
    float4 v = ((const float4*)src)[i];
    uint2 u; u.x = pk2bf(v.x, v.y); u.y = pk2bf(v.z, v.w);
    ((uint2*)dst)[i] = u;
  }
}

// ------------------------------------------------------------- WMMA GEMM
// C[M,N] = A[M,K](bf16) * W[N,K](bf16)^T  (+bias)(+accum f32)(+gelu)
// Staging: async global->LDS (ASYNCcnt), double-buffered, 8 waves/WG.
#define BM   128
#define BK   32
#define KPAD 40   // shorts/row: 80B rows -> 16B aligned, bank-spread

template <int BN_, int SUBN>   // wave tile = 32 x (SUBN*16); 4x2 wave grid
__global__ __launch_bounds__(256)
void gemm_bf16_kernel(const unsigned short* __restrict__ A, int strideA, int lda,
                      const unsigned short* __restrict__ Bw, int strideB, int ldb,
                      const float* __restrict__ bias,
                      float* __restrict__ Cf, unsigned short* __restrict__ Ch,
                      int strideC, int ldc, int K, int flags) {
  __shared__ __align__(16) unsigned short As[2][BM * KPAD];
  __shared__ __align__(16) unsigned short Bs[2][BN_ * KPAD];
  constexpr int BITER = BN_ / 64;

  const int tid   = threadIdx.x;
  const int lane  = tid & 31;
  const int wid   = tid >> 5;
  const int waveM = wid >> 1;
  const int waveN = wid & 1;
  const int g     = lane >> 4;
  const int idx   = lane & 15;

  const int blockN = blockIdx.x * BN_;
  const int blockM = blockIdx.y * BM;
  const int batch  = blockIdx.z;

  const unsigned short* Ab = A  + batch * strideA;
  const unsigned short* Bb = Bw + batch * strideB;

  // each thread stages 16B chunks
  const int rA = tid >> 2;            // 0..63 (+64 on 2nd iter)
  const int cc = (tid & 3) * 8;       // k-offset within BK

  // LDS byte offsets (generic LDS pointers keep offset in low 32 bits)
  const unsigned asb = (unsigned)(unsigned long long)(void*)&As[0][0];
  const unsigned bsb = (unsigned)(unsigned long long)(void*)&Bs[0][0];
  const unsigned aOff = (unsigned)((rA * KPAD + cc) * 2);
  const unsigned bufA = (unsigned)(BM * KPAD * 2);
  const unsigned bufB = (unsigned)(BN_ * KPAD * 2);

  v8f acc[2][SUBN] = {};

#if USE_ASYNC_STAGE
  auto stageAsync = [&](int k0, int p) {
#pragma unroll
    for (int i = 0; i < 2; ++i) {
      unsigned lds = asb + p * bufA + aOff + (unsigned)(i * 64 * KPAD * 2);
      unsigned long long ga =
          (unsigned long long)(const void*)(Ab + (blockM + rA + i * 64) * lda + k0 + cc);
      asm volatile("global_load_async_to_lds_b128 %0, %1, off"
                   :: "v"(lds), "v"(ga) : "memory");
    }
#pragma unroll
    for (int i = 0; i < BITER; ++i) {
      unsigned lds = bsb + p * bufB + aOff + (unsigned)(i * 64 * KPAD * 2);
      unsigned long long ga =
          (unsigned long long)(const void*)(Bb + (blockN + rA + i * 64) * ldb + k0 + cc);
      asm volatile("global_load_async_to_lds_b128 %0, %1, off"
                   :: "v"(lds), "v"(ga) : "memory");
    }
  };
#else
  uint4 aR[2], bR[BITER];
  auto loadTiles = [&](int k0) {
#pragma unroll
    for (int i = 0; i < 2; ++i)
      aR[i] = *(const uint4*)(Ab + (blockM + rA + i * 64) * lda + k0 + cc);
#pragma unroll
    for (int i = 0; i < BITER; ++i)
      bR[i] = *(const uint4*)(Bb + (blockN + rA + i * 64) * ldb + k0 + cc);
  };
  auto storeTiles = [&](int p) {
#pragma unroll
    for (int i = 0; i < 2; ++i)
      *(uint4*)&As[p][(rA + i * 64) * KPAD + cc] = aR[i];
#pragma unroll
    for (int i = 0; i < BITER; ++i)
      *(uint4*)&Bs[p][(rA + i * 64) * KPAD + cc] = bR[i];
  };
#endif

#if USE_ASYNC_STAGE
  stageAsync(0, 0);
  asm volatile("s_wait_asynccnt 0" ::: "memory");
  __syncthreads();
#else
  loadTiles(0); storeTiles(0); __syncthreads();
#endif

  const int nk = K / BK;
  for (int kk = 0; kk < nk; ++kk) {
    const int p = kk & 1;
#if USE_ASYNC_STAGE
    if (kk + 1 < nk) stageAsync((kk + 1) * BK, p ^ 1);   // overlaps with WMMA below
#else
    if (kk + 1 < nk) loadTiles((kk + 1) * BK);
#endif

    // batch-load all fragments so s_wait_dscnt staggers across the WMMA burst
    // (ISA 16-bit layout: lane<16 K{0..7,16..23}; lane>=16 K{8..15,24..31})
    v16bf afr[2], bfr[SUBN];
#pragma unroll
    for (int ti = 0; ti < 2; ++ti) {
      int row = waveM * 32 + ti * 16 + idx;
      v16us t;
      ((uint4*)&t)[0] = *(const uint4*)&As[p][row * KPAD + 8 * g];
      ((uint4*)&t)[1] = *(const uint4*)&As[p][row * KPAD + 16 + 8 * g];
      afr[ti] = __builtin_bit_cast(v16bf, t);
    }
#pragma unroll
    for (int tj = 0; tj < SUBN; ++tj) {
      int row = waveN * (SUBN * 16) + tj * 16 + idx;
      v16us t;
      ((uint4*)&t)[0] = *(const uint4*)&Bs[p][row * KPAD + 8 * g];
      ((uint4*)&t)[1] = *(const uint4*)&Bs[p][row * KPAD + 16 + 8 * g];
      bfr[tj] = __builtin_bit_cast(v16bf, t);
    }
#pragma unroll
    for (int ti = 0; ti < 2; ++ti)
#pragma unroll
      for (int tj = 0; tj < SUBN; ++tj)
        acc[ti][tj] = __builtin_amdgcn_wmma_f32_16x16x32_bf16(
            false, afr[ti], false, bfr[tj], (short)0, acc[ti][tj], false, false);

#if USE_ASYNC_STAGE
    if (kk + 1 < nk) asm volatile("s_wait_asynccnt 0" ::: "memory");
#else
    if (kk + 1 < nk) storeTiles(p ^ 1);
#endif
    __syncthreads();
  }

  const bool do_accum = (flags & 1) != 0;
  const bool do_gelu  = (flags & 2) != 0;
  float*          Cfb = Cf ? Cf + batch * strideC : nullptr;
  unsigned short* Chb = Ch ? Ch + batch * strideC : nullptr;
#pragma unroll
  for (int ti = 0; ti < 2; ++ti) {
#pragma unroll
    for (int tj = 0; tj < SUBN; ++tj) {
      int n = blockN + waveN * (SUBN * 16) + tj * 16 + idx;
      float bn = bias ? bias[n] : 0.0f;
      int mb = blockM + waveM * 32 + ti * 16 + 8 * g;
#pragma unroll
      for (int j = 0; j < 8; ++j) {
        int off = (mb + j) * ldc + n;
        float r = acc[ti][tj][j] + bn;
        if (do_accum) r += Cfb[off];
        if (do_gelu) {
          float z = 1.5957691216057308f * (r + 0.044715f * r * r * r);
          r = r / (1.0f + __expf(-z));      // r * sigmoid == gelu_tanh
        }
        if (Cfb) Cfb[off] = r;
        if (Chb) Chb[off] = f2bf(r);
      }
    }
  }
}

// ------------------------------------------------------------- softmax (f32 in, bf16 out)
__global__ __launch_bounds__(256)
void softmax_kernel(const float* __restrict__ s, unsigned short* __restrict__ pbf,
                    int T, int Scols, int causal) {
  __shared__ float red[256];
  int row = blockIdx.x;           // bh*T + t
  int t   = row % T;
  const float* p = s + row * Scols;
  unsigned short* q = pbf + row * Scols;
  int valid = causal ? (t + 1) : Scols;
  int tid = threadIdx.x;

  float lmax = -3.0e38f;
  for (int i = tid; i < Scols; i += 256)
    lmax = fmaxf(lmax, (i < valid) ? p[i] : -3.0e38f);
  red[tid] = lmax; __syncthreads();
  for (int st = 128; st > 0; st >>= 1) {
    if (tid < st) red[tid] = fmaxf(red[tid], red[tid + st]);
    __syncthreads();
  }
  float m = red[0]; __syncthreads();

  float lsum = 0.0f;
  float ev[2];
#pragma unroll
  for (int i = 0; i < 2; ++i) {
    int c = tid + i * 256;
    float v = (c < valid) ? __expf(p[c] - m) : 0.0f;
    ev[i] = v; lsum += v;
  }
  red[tid] = lsum; __syncthreads();
  for (int st = 128; st > 0; st >>= 1) {
    if (tid < st) red[tid] += red[tid + st];
    __syncthreads();
  }
  float inv = 1.0f / red[0];
#pragma unroll
  for (int i = 0; i < 2; ++i) q[tid + i * 256] = f2bf(ev[i] * inv);
}

// ------------------------------------------------------------- embed + LN
__global__ __launch_bounds__(256)
void embed_ln_kernel(const int* __restrict__ ids, const float* __restrict__ tok,
                     const float* __restrict__ pos, const float* __restrict__ w,
                     const float* __restrict__ b, float* __restrict__ x,
                     unsigned short* __restrict__ xb) {
  __shared__ float r1[256], r2[256];
  int row = blockIdx.x;            // t*B + b
  int t = row / B_, bb = row % B_;
  int id = ids[bb * T_ + t];
  const float* ts = tok + (long long)id * D_;
  const float* ps = pos + (t + POFF_) * D_;
  int tid = threadIdx.x;
  float vals[4]; float s = 0.0f, sq = 0.0f;
#pragma unroll
  for (int i = 0; i < 4; ++i) {
    int d = tid + i * 256;
    float v = ts[d] + ps[d];
    vals[i] = v; s += v; sq += v * v;
  }
  r1[tid] = s; r2[tid] = sq; __syncthreads();
  for (int st = 128; st > 0; st >>= 1) {
    if (tid < st) { r1[tid] += r1[tid + st]; r2[tid] += r2[tid + st]; }
    __syncthreads();
  }
  float mean = r1[0] * (1.0f / D_);
  float var  = r2[0] * (1.0f / D_) - mean * mean;
  float inv  = rsqrtf(var + EPS_);
#pragma unroll
  for (int i = 0; i < 4; ++i) {
    int d = tid + i * 256;
    float o = (vals[i] - mean) * inv * w[d] + b[d];
    x[row * D_ + d]  = o;
    xb[row * D_ + d] = f2bf(o);
  }
}

// ------------------------------------------------------------- add + LN
__global__ __launch_bounds__(256)
void add_ln_kernel(const float* __restrict__ res, const float* __restrict__ y,
                   const float* __restrict__ w, const float* __restrict__ b,
                   float* __restrict__ x, unsigned short* __restrict__ xb) {
  __shared__ float r1[256], r2[256];
  int base = blockIdx.x * D_;
  int tid = threadIdx.x;
  float vals[4]; float s = 0.0f, sq = 0.0f;
#pragma unroll
  for (int i = 0; i < 4; ++i) {
    int d = tid + i * 256;
    float v = res[base + d] + y[base + d];
    vals[i] = v; s += v; sq += v * v;
  }
  r1[tid] = s; r2[tid] = sq; __syncthreads();
  for (int st = 128; st > 0; st >>= 1) {
    if (tid < st) { r1[tid] += r1[tid + st]; r2[tid] += r2[tid + st]; }
    __syncthreads();
  }
  float mean = r1[0] * (1.0f / D_);
  float var  = r2[0] * (1.0f / D_) - mean * mean;
  float inv  = rsqrtf(var + EPS_);
#pragma unroll
  for (int i = 0; i < 4; ++i) {
    int d = tid + i * 256;
    float o = (vals[i] - mean) * inv * w[d] + b[d];
    x[base + d]  = o;
    xb[base + d] = f2bf(o);
  }
}

// ------------------------------------------------------------- head packers
__global__ void pack_rows_kernel(const float* __restrict__ src, int ldsrc, int col0,
                                 unsigned short* __restrict__ dst, float scale) {
  int idx = blockIdx.x * 256 + threadIdx.x;   // bh*T*HD
  int hd = idx % HD_;
  int t  = (idx / HD_) % T_;
  int bh = idx / (HD_ * T_);
  int b = bh / H_, h = bh % H_;
  dst[idx] = f2bf(src[(t * B_ + b) * ldsrc + col0 + h * HD_ + hd] * scale);
}

__global__ void pack_vt_kernel(const float* __restrict__ src, int ldsrc, int col0,
                               unsigned short* __restrict__ dst) {  // [BH, HD, S]
  int idx = blockIdx.x * 256 + threadIdx.x;
  int s  = idx % S_;
  int hd = (idx / S_) % HD_;
  int bh = idx / (S_ * HD_);
  int b = bh / H_, h = bh % H_;
  dst[idx] = f2bf(src[(s * B_ + b) * ldsrc + col0 + h * HD_ + hd]);
}

__global__ void unpack_attn_kernel(const float* __restrict__ attno,
                                   unsigned short* __restrict__ aob) {
  int idx = blockIdx.x * 256 + threadIdx.x;   // m*D + d
  int d = idx % D_;
  int m = idx / D_;
  int t = m / B_, b = m % B_;
  int h = d / HD_, hd = d % HD_;
  aob[idx] = f2bf(attno[((b * H_ + h) * T_ + t) * HD_ + hd]);
}

__global__ void out_transpose_kernel(const float* __restrict__ x, float* __restrict__ out) {
  int idx = blockIdx.x * 256 + threadIdx.x;   // b*T*D + t*D + d
  int d = idx % D_;
  int t = (idx / D_) % T_;
  int b = idx / (D_ * T_);
  out[idx] = x[(t * B_ + b) * D_ + d];
}

// ------------------------------------------------------------- launcher
extern "C" void kernel_launch(void* const* d_in, const int* in_sizes, int n_in,
                              void* d_out, int out_size, void* d_ws, size_t ws_size,
                              hipStream_t stream) {
  (void)in_sizes; (void)n_in; (void)out_size; (void)ws_size;

  const int*   ids   = (const int*)d_in[0];
  const float* enc   = (const float*)d_in[1];
  const float* etok  = (const float*)d_in[2];
  const float* epos  = (const float*)d_in[3];
  const float* lnew  = (const float*)d_in[4];
  const float* lneb  = (const float*)d_in[5];
  const float* sa_w  = (const float*)d_in[6];
  const float* sa_b  = (const float*)d_in[7];
  const float* ca_w  = (const float*)d_in[8];
  const float* ca_b  = (const float*)d_in[9];
  const float* wat_w = (const float*)d_in[10];
  const float* wat_b = (const float*)d_in[11];
  const float* fc1w  = (const float*)d_in[12];
  const float* fc1b  = (const float*)d_in[13];
  const float* fc2w  = (const float*)d_in[14];
  const float* fc2b  = (const float*)d_in[15];
  const float* lnw   = (const float*)d_in[16];
  const float* lnb   = (const float*)d_in[17];
  float* out = (float*)d_out;

  size_t woff = 0;
  auto carve = [&](size_t bytes) -> void* {
    void* p = (char*)d_ws + woff;
    woff += (bytes + 255) & ~(size_t)255;
    return p;
  };
  float* x     = (float*)carve(sizeof(float) * M_ * D_);
  float* res   = (float*)carve(sizeof(float) * M_ * D_);
  float* tmp1  = (float*)carve(sizeof(float) * M_ * FF_);
  float* tmp2  = (float*)carve(sizeof(float) * M_ * D_);
  float* accb  = (float*)carve(sizeof(float) * M_ * D_);
  float* attno = (float*)carve(sizeof(float) * BH_ * T_ * HD_);
  float* sc    = (float*)carve(sizeof(float) * (size_t)BH_ * T_ * S_);
  unsigned short* xb   = (unsigned short*)carve(2ull * M_ * D_);
  unsigned short* encb = (unsigned short*)carve(2ull * 3 * S_ * B_ * D_);
  unsigned short* qhb  = (unsigned short*)carve(2ull * BH_ * T_ * HD_);
  unsigned short* khb  = (unsigned short*)carve(2ull * BH_ * S_ * HD_);
  unsigned short* vtb  = (unsigned short*)carve(2ull * BH_ * HD_ * S_);
  unsigned short* pb   = (unsigned short*)carve(2ull * BH_ * T_ * S_);
  unsigned short* aob  = (unsigned short*)carve(2ull * M_ * D_);
  unsigned short* t3b  = (unsigned short*)carve(2ull * M_ * D_);
  unsigned short* hb   = (unsigned short*)carve(2ull * M_ * FF_);
  unsigned short* wb_sa  = (unsigned short*)carve(2ull * 4 * D_ * D_);
  unsigned short* wb_ca  = (unsigned short*)carve(2ull * 12 * D_ * D_);
  unsigned short* wb_wat = (unsigned short*)carve(2ull * 3 * D_ * D_);
  unsigned short* wb_f1  = (unsigned short*)carve(2ull * FF_ * D_);
  unsigned short* wb_f2  = (unsigned short*)carve(2ull * D_ * FF_);

  auto cvt = [&](const float* s, unsigned short* d, long long n) {
    cvt_bf16_kernel<<<(int)(n / 4 / 256), 256, 0, stream>>>(s, d, (int)(n / 4));
  };
  auto gemm128 = [&](const unsigned short* A, int sA, int lda,
                     const unsigned short* Bp, int sB, int ldb,
                     const float* bias, float* Cf, unsigned short* Ch,
                     int sC, int ldc, int M, int N, int K, int batch, int flags) {
    dim3 grid(N / 128, M / BM, batch);
    gemm_bf16_kernel<128, 4><<<grid, dim3(256), 0, stream>>>(
        A, sA, lda, Bp, sB, ldb, bias, Cf, Ch, sC, ldc, K, flags);
  };
  auto gemm64 = [&](const unsigned short* A, int sA, int lda,
                    const unsigned short* Bp, int sB, int ldb,
                    const float* bias, float* Cf, unsigned short* Ch,
                    int sC, int ldc, int M, int N, int K, int batch, int flags) {
    dim3 grid(N / 64, M / BM, batch);
    gemm_bf16_kernel<64, 2><<<grid, dim3(256), 0, stream>>>(
        A, sA, lda, Bp, sB, ldb, bias, Cf, Ch, sC, ldc, K, flags);
  };

  auto attention = [&](int causal) {  // qhb/khb/vtb -> aob
    gemm128(qhb, T_ * HD_, HD_, khb, S_ * HD_, HD_, nullptr,
            sc, nullptr, T_ * S_, S_, T_, S_, HD_, BH_, 0);
    softmax_kernel<<<BH_ * T_, 256, 0, stream>>>(sc, pb, T_, S_, causal);
    gemm64(pb, T_ * S_, S_, vtb, HD_ * S_, S_, nullptr,
           attno, nullptr, T_ * HD_, HD_, T_, HD_, S_, BH_, 0);
    unpack_attn_kernel<<<(M_ * D_) / 256, 256, 0, stream>>>(attno, aob);
  };

  const int PK = (BH_ * T_ * HD_) / 256;
  const size_t XB = sizeof(float) * (size_t)M_ * D_;

  cvt(enc, encb, 3LL * S_ * B_ * D_);
  embed_ln_kernel<<<M_, 256, 0, stream>>>(ids, etok, epos, lnew, lneb, x, xb);

  for (int l = 0; l < L_; ++l) {
    cvt(sa_w  + (long long)l * 4 * D_ * D_,  wb_sa,  4LL * D_ * D_);
    cvt(ca_w  + (long long)l * 12 * D_ * D_, wb_ca,  12LL * D_ * D_);
    cvt(wat_w + (long long)l * 3 * D_ * D_,  wb_wat, 3LL * D_ * D_);
    cvt(fc1w  + (long long)l * FF_ * D_,     wb_f1,  (long long)FF_ * D_);
    cvt(fc2w  + (long long)l * D_ * FF_,     wb_f2,  (long long)D_ * FF_);

    const float* sab = sa_b + l * 4 * D_;

    // ---- self attention (fused QKV) ----
    hipMemcpyAsync(res, x, XB, hipMemcpyDeviceToDevice, stream);
    gemm128(xb, 0, D_, wb_sa, 0, D_, sab, tmp1, nullptr, 0, 3 * D_, M_, 3 * D_, D_, 1, 0);
    pack_rows_kernel<<<PK, 256, 0, stream>>>(tmp1, 3 * D_, 0, qhb, 0.125f);
    pack_rows_kernel<<<PK, 256, 0, stream>>>(tmp1, 3 * D_, D_, khb, 1.0f);
    pack_vt_kernel<<<PK, 256, 0, stream>>>(tmp1, 3 * D_, 2 * D_, vtb);
    attention(1);
    gemm128(aob, 0, D_, wb_sa + 3LL * D_ * D_, 0, D_, sab + 3 * D_,
            tmp2, nullptr, 0, D_, M_, D_, D_, 1, 0);
    add_ln_kernel<<<M_, 256, 0, stream>>>(res, tmp2, lnw + (l * 3 + 0) * D_,
                                          lnb + (l * 3 + 0) * D_, x, xb);

    // ---- 3x cross attention, concat folded into accumulated wattn GEMM ----
    hipMemcpyAsync(res, x, XB, hipMemcpyDeviceToDevice, stream);
    for (int i = 0; i < 3; ++i) {
      const unsigned short* cw = wb_ca + (long long)i * 4 * D_ * D_;
      const float*          cb = ca_b + ((long long)l * 3 + i) * 4 * D_;
      gemm128(xb, 0, D_, cw, 0, D_, cb, tmp2, nullptr, 0, D_, M_, D_, D_, 1, 0);   // Q
      gemm128(encb + (long long)i * S_ * B_ * D_, 0, D_,
              cw + (long long)D_ * D_, 0, D_, cb + D_,
              tmp1, nullptr, 0, 2 * D_, M_, 2 * D_, D_, 1, 0);                      // K,V
      pack_rows_kernel<<<PK, 256, 0, stream>>>(tmp2, D_, 0, qhb, 0.125f);
      pack_rows_kernel<<<PK, 256, 0, stream>>>(tmp1, 2 * D_, 0, khb, 1.0f);
      pack_vt_kernel<<<PK, 256, 0, stream>>>(tmp1, 2 * D_, D_, vtb);
      attention(0);
      gemm128(aob, 0, D_, cw + 3LL * D_ * D_, 0, D_, cb + 3 * D_,
              nullptr, t3b, 0, D_, M_, D_, D_, 1, 0);
      gemm128(t3b, 0, D_, wb_wat + i * D_, 0, 3 * D_,
              (i == 0) ? (wat_b + l * D_) : nullptr,
              accb, nullptr, 0, D_, M_, D_, D_, 1, (i == 0) ? 0 : 1);
    }
    add_ln_kernel<<<M_, 256, 0, stream>>>(res, accb, lnw + (l * 3 + 1) * D_,
                                          lnb + (l * 3 + 1) * D_, x, xb);

    // ---- FFN ----
    hipMemcpyAsync(res, x, XB, hipMemcpyDeviceToDevice, stream);
    gemm128(xb, 0, D_, wb_f1, 0, D_, fc1b + l * FF_,
            nullptr, hb, 0, FF_, M_, FF_, D_, 1, 2 /*gelu -> bf16*/);
    gemm128(hb, 0, FF_, wb_f2, 0, FF_, fc2b + l * D_,
            tmp2, nullptr, 0, D_, M_, D_, FF_, 1, 0);
    add_ln_kernel<<<M_, 256, 0, stream>>>(res, tmp2, lnw + (l * 3 + 2) * D_,
                                          lnb + (l * 3 + 2) * D_, x, xb);
  }

  out_transpose_kernel<<<(B_ * T_ * D_) / 256, 256, 0, stream>>>(x, out);
}
